// CDDModel_69337952027170
// MI455X (gfx1250) — matrix operand order repaired and checked by reading, not compiled, not run on hardware
//
#include <hip/hip_runtime.h>
#include <hip/hip_bf16.h>
#include <math.h>

// ---------------- types ----------------
typedef _Float16 h8   __attribute__((ext_vector_type(8)));
typedef _Float16 v16h __attribute__((ext_vector_type(16)));
typedef float    v8f  __attribute__((ext_vector_type(8)));

union F16x16 { v16h v; h8 h[2]; };

// A fragment: row-major [16][Kpad] f16 tile. lane<16: row=lane, K {0..7,16..23}+k0 ;
// lane>=16: row=lane-16, K {8..15,24..31}+k0 (CDNA5 16-bit A layout).
static __device__ inline v16h load_a_frag(const _Float16* A, int Kpad, int k0, int lane) {
  int row = lane & 15;
  int kb  = (lane < 16) ? 0 : 8;
  const _Float16* p = A + row * Kpad + k0 + kb;
  F16x16 f;
  f.h[0] = *(const h8*)(p);
  f.h[1] = *(const h8*)(p + 16);
  return f.v;
}
// B fragment from weights stored [N][Kpad] (i.e. natural [out][in]):
// lane<16: col=lane, K {0..7,16..23} ; lane>=16: col=lane-16, K {8..15,24..31} (matches A).
static __device__ inline v16h load_b_frag(const _Float16* B, int Kpad, int k0, int lane) {
  int col = lane & 15;
  int kb  = (lane < 16) ? 0 : 8;
  const _Float16* p = B + col * Kpad + k0 + kb;
  F16x16 f;
  f.h[0] = *(const h8*)(p);
  f.h[1] = *(const h8*)(p + 16);
  return f.v;
}

// wave-level GEMM: out[16][ostride] (f32) = A[16][Kpad] * W^T + bias, W=[ntiles*16][Kpad] f16
static __device__ inline void gemm_store(const _Float16* A, int Kpad, const _Float16* W,
                                         const float* bias, int nvalid, int ntiles,
                                         float* out, int ostride, int lane) {
  const int nl = lane & 15;
  const int mb = (lane < 16) ? 0 : 8;
  for (int nt = 0; nt < ntiles; ++nt) {
    v8f c = {};
    const _Float16* B = W + (size_t)(nt * 16) * Kpad;
    for (int k = 0; k < Kpad; k += 32) {
      v16h a = load_a_frag(A, Kpad, k, lane);
      v16h b = load_b_frag(B, Kpad, k, lane);
      c = __builtin_amdgcn_wmma_f32_16x16x32_f16(false, a, false, b, (short)0, c, false, false);
    }
    int n = nt * 16 + nl;
    float bv = (n < nvalid) ? bias[n] : 0.f;
    for (int r = 0; r < 8; ++r) out[(r + mb) * ostride + n] = c[r] + bv;
  }
  __syncthreads();
}

// build f16 A tile: [s (svalid, optional relu) | vector-norms (H) | zero pad]
static __device__ inline void build_X(_Float16* Xa, int Kpad, const float* sb, int sstride,
                                      int svalid, bool relu, const float* vhb, int sin, int H,
                                      int lane) {
  for (int t = lane; t < 16 * Kpad; t += 32) {
    int e = t / Kpad, k = t % Kpad;
    float x = 0.f;
    if (k < svalid) { x = sb[e * sstride + k]; if (relu) x = fmaxf(x, 0.f); }
    else if (k < svalid + H) {
      int hh = k - svalid;
      float a0 = vhb[e * sin + hh * 3 + 0];
      float a1 = vhb[e * sin + hh * 3 + 1];
      float a2 = vhb[e * sin + hh * 3 + 2];
      x = sqrtf(fmaxf(a0 * a0 + a1 * a1 + a2 * a2, 1e-8f));
    }
    Xa[t] = (_Float16)x;
  }
  __syncthreads();
}

// vh[e][h][c] = sum_v wh[h][v] * vin[e][v][c]
static __device__ inline void vh_stage(const float* wh, int Hout, int Hin, const float* vin,
                                       int sin, float* vout_, int sout, int lane) {
  for (int t = lane; t < 16 * Hout * 3; t += 32) {
    int e = t / (Hout * 3), r = t % (Hout * 3), hh = r / 3, c = r % 3;
    float acc = 0.f;
    for (int vv = 0; vv < Hin; ++vv) acc += wh[hh * Hin + vv] * vin[e * sin + vv * 3 + c];
    vout_[e * sout + hh * 3 + c] = acc;
  }
  __syncthreads();
}

// v_out[e][o][c] = gate[e][o] * sum_h wv[o][h]*vh[e][h][c]
static __device__ inline void v_out_stage(const float* wv, int vo, int Hin, const float* vhb,
                                          int sin, const float* gbuf, int gstride, float* out,
                                          int sout, int lane) {
  for (int t = lane; t < 16 * vo * 3; t += 32) {
    int e = t / (vo * 3), r = t % (vo * 3), o = r / 3, c = r % 3;
    float acc = 0.f;
    for (int hh = 0; hh < Hin; ++hh) acc += wv[o * Hin + hh] * vhb[e * sin + hh * 3 + c];
    out[e * sout + o * 3 + c] = acc * gbuf[e * gstride + o];
  }
  __syncthreads();
}

static __device__ inline void sigmoid_stage(float* g, int n, int lane) {
  for (int t = lane; t < n; t += 32) g[t] = 1.f / (1.f + expf(-g[t]));
  __syncthreads();
}

// in-place LayerNorm over 100 dims, one node per lane (lanes 0..15)
static __device__ inline void ln_stage(float* s, const float* g, const float* b, int lane) {
  if (lane < 16) {
    float mu = 0.f;
    for (int k = 0; k < 100; ++k) mu += s[lane * 100 + k];
    mu *= 0.01f;
    float var = 0.f;
    for (int k = 0; k < 100; ++k) { float d = s[lane * 100 + k] - mu; var += d * d; }
    var *= 0.01f;
    float inv = rsqrtf(var + 1e-5f);
    for (int k = 0; k < 100; ++k)
      s[lane * 100 + k] = (s[lane * 100 + k] - mu) * inv * g[k] + b[k];
  }
  __syncthreads();
}

// v /= sqrt(mean_over_16_vectors(max(||v||^2, eps)))
static __device__ inline void vscale_stage(float* v, int lane) {
  if (lane < 16) {
    float acc = 0.f;
    for (int o = 0; o < 16; ++o) {
      float a0 = v[lane * 48 + o * 3 + 0];
      float a1 = v[lane * 48 + o * 3 + 1];
      float a2 = v[lane * 48 + o * 3 + 2];
      acc += fmaxf(a0 * a0 + a1 * a1 + a2 * a2, 1e-8f);
    }
    float inv = rsqrtf(acc * (1.f / 16.f));
    for (int r = 0; r < 48; ++r) v[lane * 48 + r] *= inv;
  }
  __syncthreads();
}

// ---------------- small kernels ----------------
__global__ void k_pack(const float* src, _Float16* dst, int rs, int cs, int rd, int cd) {
  int t = blockIdx.x * blockDim.x + threadIdx.x;
  if (t >= rd * cd) return;
  int r = t / cd, c = t % cd;
  dst[t] = (_Float16)((r < rs && c < cs) ? src[r * cs + c] : 0.f);
}

__global__ void k_count(const int* idx, float* cnt, int n) {
  int i = blockIdx.x * blockDim.x + threadIdx.x;
  if (i < n) atomicAdd(&cnt[idx[i]], 1.f);
}

__global__ void k_node_init(const int* atoms, const float* embed, const float* lnb,
                            const float* lng, const float* W, const float* b, float* hs,
                            float* hv, int N) {
  int i = blockIdx.x * blockDim.x + threadIdx.x;
  if (i >= N) return;
  float e[13];
  int at = atoms[i];
  float mu = 0.f;
  for (int k = 0; k < 13; ++k) { e[k] = embed[at * 13 + k]; mu += e[k]; }
  mu *= (1.f / 13.f);
  float var = 0.f;
  for (int k = 0; k < 13; ++k) { float d = e[k] - mu; var += d * d; }
  var *= (1.f / 13.f);
  float inv = rsqrtf(var + 1e-5f);
  for (int k = 0; k < 13; ++k) e[k] = (e[k] - mu) * inv * lng[k] + lnb[k];
  for (int o = 0; o < 100; ++o) {
    float acc = b[o];
    for (int k = 0; k < 13; ++k) acc += W[o * 13 + k] * e[k];
    hs[i * 100 + o] = acc;
  }
  for (int r = 0; r < 48; ++r) hv[i * 48 + r] = 0.f;
}

__global__ void k_edge_init(const float* edge_s, const float* edge_v, const float* lnb,
                            const float* lng, const float* wh, const float* wsb,
                            const float* wsw, const float* wsvb, const float* wsvw,
                            const float* wv, float* es, float* ev, int E) {
  int i = blockIdx.x * blockDim.x + threadIdx.x;
  if (i >= E) return;
  float s[17];
  float mu = 0.f;
  for (int k = 0; k < 16; ++k) { s[k] = edge_s[i * 16 + k]; mu += s[k]; }
  mu *= (1.f / 16.f);
  float var = 0.f;
  for (int k = 0; k < 16; ++k) { float d = s[k] - mu; var += d * d; }
  var *= (1.f / 16.f);
  float inv = rsqrtf(var + 1e-5f);
  for (int k = 0; k < 16; ++k) s[k] = (s[k] - mu) * inv * lng[k] + lnb[k];
  float vx = edge_v[i * 3 + 0], vy = edge_v[i * 3 + 1], vz = edge_v[i * 3 + 2];
  float n2 = fmaxf(vx * vx + vy * vy + vz * vz, 1e-8f);
  float vinv = rsqrtf(n2);
  vx *= vinv; vy *= vinv; vz *= vinv;
  float w = wh[0];
  float hx = w * vx, hy = w * vy, hz = w * vz;
  s[16] = sqrtf(fmaxf(hx * hx + hy * hy + hz * hz, 1e-8f));
  float so[32];
  for (int o = 0; o < 32; ++o) {
    float acc = wsb[o];
    for (int k = 0; k < 17; ++k) acc += wsw[o * 17 + k] * s[k];
    so[o] = acc;
    es[i * 32 + o] = acc;
  }
  float g = wsvb[0];
  for (int o = 0; o < 32; ++o) g += wsvw[o] * so[o];
  g = 1.f / (1.f + expf(-g));
  float ww = wv[0];
  ev[i * 3 + 0] = ww * hx * g;
  ev[i * 3 + 1] = ww * hy * g;
  ev[i * 3 + 2] = ww * hz * g;
}

// ---------------- conv (edge message) kernel: 1 wave = 16 edges ----------------
struct ConvArgs {
  const float *s, *v, *es, *ev;
  const int *src, *dst;
  const _Float16* W0; const float *b0, *wh0, *wv0; const _Float16* G0; const float* gb0;
  const _Float16* W1; const float *b1, *wh1, *wv1; const _Float16* G1; const float* gb1;
  const _Float16* W2; const float *b2, *wh2, *wv2; const _Float16* G2; const float* gb2;
  float *dhs, *dhv;
  int E;
};

__global__ __launch_bounds__(32) void k_conv(ConvArgs a) {
  __shared__ alignas(16) _Float16 Xa[16 * 288];
  __shared__ float sbuf[16 * 112];
  __shared__ float vA[16 * 99];
  __shared__ float vB[16 * 99];
  __shared__ float gbuf[16 * 16];
  __shared__ int sid[16], did[16];
  const int lane = threadIdx.x;
  const int e0 = blockIdx.x * 16;
  const int evalid = (a.E - e0 < 16) ? (a.E - e0) : 16;
  if (lane < 16) {
    int ee = e0 + lane; if (ee >= a.E) ee = a.E - 1;
    sid[lane] = a.src[ee];
    did[lane] = a.dst[ee];
  }
  __syncthreads();
  // X0 scalars: [s[src](100) | es(32) | s[dst](100) | vn(33, later) | pad]
  for (int t = lane; t < 16 * 288; t += 32) {
    int e = t / 288, k = t % 288;
    int ge = e0 + e; if (ge >= a.E) ge = a.E - 1;
    float x = 0.f;
    if (k < 100)       x = a.s[sid[e] * 100 + k];
    else if (k < 132)  x = a.es[(size_t)ge * 32 + (k - 100)];
    else if (k < 232)  x = a.s[did[e] * 100 + (k - 132)];
    Xa[t] = (_Float16)x;
  }
  // mv = [v[src](16) | ev(1) | v[dst](16)]
  for (int t = lane; t < 16 * 99; t += 32) {
    int e = t / 99, r = t % 99, ch = r / 3, c = r % 3;
    int ge = e0 + e; if (ge >= a.E) ge = a.E - 1;
    float x;
    if (ch < 16)       x = a.v[sid[e] * 48 + ch * 3 + c];
    else if (ch == 16) x = a.ev[(size_t)ge * 3 + c];
    else               x = a.v[did[e] * 48 + (ch - 17) * 3 + c];
    vA[t] = x;
  }
  __syncthreads();
  // vh0 = wh0(33x33) @ mv -> vB
  for (int t = lane; t < 16 * 99; t += 32) {
    int e = t / 99, r = t % 99, hh = r / 3, c = r % 3;
    float acc = 0.f;
    for (int vv = 0; vv < 33; ++vv) acc += a.wh0[hh * 33 + vv] * vA[e * 99 + vv * 3 + c];
    vB[t] = acc;
  }
  __syncthreads();
  for (int t = lane; t < 16 * 33; t += 32) {
    int e = t / 33, hh = t % 33;
    float x0 = vB[e * 99 + hh * 3], x1 = vB[e * 99 + hh * 3 + 1], x2 = vB[e * 99 + hh * 3 + 2];
    Xa[e * 288 + 232 + hh] = (_Float16)sqrtf(fmaxf(x0 * x0 + x1 * x1 + x2 * x2, 1e-8f));
  }
  __syncthreads();
  // ---- m0 ----
  gemm_store(Xa, 288, a.W0, a.b0, 100, 7, sbuf, 112, lane);
  build_X(Xa, 128, sbuf, 112, 100, false, (const float*)0, 0, 0, lane);
  gemm_store(Xa, 128, a.G0, a.gb0, 16, 1, gbuf, 16, lane);
  sigmoid_stage(gbuf, 16 * 16, lane);
  v_out_stage(a.wv0, 16, 33, vB, 99, gbuf, 16, vA, 48, lane);
  // ---- m1 ----
  vh_stage(a.wh1, 16, 16, vA, 48, vB, 48, lane);
  build_X(Xa, 128, sbuf, 112, 100, true, vB, 48, 16, lane);
  gemm_store(Xa, 128, a.W1, a.b1, 100, 7, sbuf, 112, lane);
  build_X(Xa, 128, sbuf, 112, 100, false, (const float*)0, 0, 0, lane);
  gemm_store(Xa, 128, a.G1, a.gb1, 16, 1, gbuf, 16, lane);
  sigmoid_stage(gbuf, 16 * 16, lane);
  v_out_stage(a.wv1, 16, 16, vB, 48, gbuf, 16, vA, 48, lane);
  // ---- m2 (no scalar act) ----
  vh_stage(a.wh2, 16, 16, vA, 48, vB, 48, lane);
  build_X(Xa, 128, sbuf, 112, 100, true, vB, 48, 16, lane);
  gemm_store(Xa, 128, a.W2, a.b2, 100, 7, sbuf, 112, lane);
  build_X(Xa, 128, sbuf, 112, 100, false, (const float*)0, 0, 0, lane);
  gemm_store(Xa, 128, a.G2, a.gb2, 16, 1, gbuf, 16, lane);
  sigmoid_stage(gbuf, 16 * 16, lane);
  v_out_stage(a.wv2, 16, 16, vB, 48, gbuf, 16, vA, 48, lane);
  // scatter (segment-sum) to dst node
  for (int t = lane; t < 16 * 100; t += 32) {
    int e = t / 100, k = t % 100;
    if (e < evalid) atomicAdd(&a.dhs[(size_t)did[e] * 100 + k], sbuf[e * 112 + k]);
  }
  for (int t = lane; t < 16 * 48; t += 32) {
    int e = t / 48, r = t % 48;
    if (e < evalid) atomicAdd(&a.dhv[(size_t)did[e] * 48 + r], vA[e * 48 + r]);
  }
}

// ---------------- node update (residual + LN + feed-forward): 1 wave = 16 nodes ----------------
struct NodeArgs {
  const float *hs_in, *hv_in, *dhs, *dhv, *deg;
  const float *n0b, *n0g, *n1b, *n1g;
  const _Float16* F0; const float *fb0, *fwh0, *fwv0; const _Float16* FG0; const float* fgb0;
  const _Float16* F1; const float *fb1, *fwh1, *fwv1; const _Float16* FG1; const float* fgb1;
  float *hs_out, *hv_out;
  int N;
};

__global__ __launch_bounds__(32) void k_node(NodeArgs a) {
  __shared__ alignas(16) _Float16 Xa[16 * 448];
  __shared__ float sbuf[16 * 400];
  __shared__ float sres[16 * 100];
  __shared__ float vres[16 * 48];
  __shared__ float vA[16 * 96];
  __shared__ float vB[16 * 96];
  __shared__ float gbuf[16 * 32];
  __shared__ float invd[16];
  const int lane = threadIdx.x;
  const int n0 = blockIdx.x * 16;
  if (lane < 16) {
    int gn = n0 + lane; if (gn >= a.N) gn = a.N - 1;
    invd[lane] = 1.f / fmaxf(a.deg[gn], 1.f);
  }
  __syncthreads();
  for (int t = lane; t < 16 * 100; t += 32) {
    int e = t / 100, k = t % 100;
    int gn = n0 + e; if (gn >= a.N) gn = a.N - 1;
    sres[t] = a.hs_in[(size_t)gn * 100 + k] + a.dhs[(size_t)gn * 100 + k] * invd[e];
  }
  for (int t = lane; t < 16 * 48; t += 32) {
    int e = t / 48, r = t % 48;
    int gn = n0 + e; if (gn >= a.N) gn = a.N - 1;
    vres[t] = a.hv_in[(size_t)gn * 48 + r] + a.dhv[(size_t)gn * 48 + r] * invd[e];
  }
  __syncthreads();
  ln_stage(sres, a.n0g, a.n0b, lane);
  vscale_stage(vres, lane);
  // ---- ff0: (100,16) -> (400,32), relu ----
  vh_stage(a.fwh0, 32, 16, vres, 48, vA, 96, lane);
  build_X(Xa, 160, sres, 100, 100, false, vA, 96, 32, lane);
  gemm_store(Xa, 160, a.F0, a.fb0, 400, 25, sbuf, 400, lane);
  build_X(Xa, 416, sbuf, 400, 400, false, (const float*)0, 0, 0, lane);
  gemm_store(Xa, 416, a.FG0, a.fgb0, 32, 2, gbuf, 32, lane);
  sigmoid_stage(gbuf, 16 * 32, lane);
  v_out_stage(a.fwv0, 32, 32, vA, 96, gbuf, 32, vB, 96, lane);
  // ---- ff1: (400,32) -> (100,16), no act ----
  vh_stage(a.fwh1, 32, 32, vB, 96, vA, 96, lane);
  build_X(Xa, 448, sbuf, 400, 400, true, vA, 96, 32, lane);
  gemm_store(Xa, 448, a.F1, a.fb1, 100, 7, sbuf, 112, lane);
  build_X(Xa, 128, sbuf, 112, 100, false, (const float*)0, 0, 0, lane);
  gemm_store(Xa, 128, a.FG1, a.fgb1, 16, 1, gbuf, 16, lane);
  sigmoid_stage(gbuf, 16 * 16, lane);
  v_out_stage(a.fwv1, 16, 32, vA, 96, gbuf, 16, vB, 48, lane);
  // residual + norm1
  for (int t = lane; t < 16 * 100; t += 32) sres[t] += sbuf[(t / 100) * 112 + (t % 100)];
  for (int t = lane; t < 16 * 48; t += 32) vres[t] += vB[t];
  __syncthreads();
  ln_stage(sres, a.n1g, a.n1b, lane);
  vscale_stage(vres, lane);
  for (int t = lane; t < 16 * 100; t += 32)
    if (n0 + t / 100 < a.N) a.hs_out[(size_t)n0 * 100 + t] = sres[t];
  for (int t = lane; t < 16 * 48; t += 32)
    if (n0 + t / 48 < a.N) a.hv_out[(size_t)n0 * 48 + t] = vres[t];
}

// ---------------- output head + pooling: 1 wave = 16 nodes ----------------
struct OutArgs {
  const float *hs, *hv, *lnb, *lng, *wh;
  const _Float16* W;
  const float* b;
  const int* bid;
  float* pooled;
  int N;
};

__global__ __launch_bounds__(32) void k_out(OutArgs a) {
  __shared__ alignas(16) _Float16 Xa[16 * 128];
  __shared__ float sres[16 * 100];
  __shared__ float vres[16 * 48];
  __shared__ float vA[16 * 48];
  __shared__ int bb[16];
  const int lane = threadIdx.x;
  const int n0 = blockIdx.x * 16;
  if (lane < 16) {
    int gn = n0 + lane; if (gn >= a.N) gn = a.N - 1;
    bb[lane] = a.bid[gn];
  }
  for (int t = lane; t < 16 * 100; t += 32) {
    int gn = n0 + t / 100; if (gn >= a.N) gn = a.N - 1;
    sres[t] = a.hs[(size_t)gn * 100 + (t % 100)];
  }
  for (int t = lane; t < 16 * 48; t += 32) {
    int gn = n0 + t / 48; if (gn >= a.N) gn = a.N - 1;
    vres[t] = a.hv[(size_t)gn * 48 + (t % 48)];
  }
  __syncthreads();
  ln_stage(sres, a.lng, a.lnb, lane);
  vscale_stage(vres, lane);
  vh_stage(a.wh, 16, 16, vres, 48, vA, 48, lane);
  build_X(Xa, 128, sres, 100, 100, false, vA, 48, 16, lane);
  const int nl = lane & 15;
  const int mb = (lane < 16) ? 0 : 8;
  for (int nt = 0; nt < 32; ++nt) {
    v8f c = {};
    const _Float16* B = a.W + (size_t)(nt * 16) * 128;
    for (int k = 0; k < 128; k += 32) {
      v16h av = load_a_frag(Xa, 128, k, lane);
      v16h bv = load_b_frag(B, 128, k, lane);
      c = __builtin_amdgcn_wmma_f32_16x16x32_f16(false, av, false, bv, (short)0, c, false, false);
    }
    int n = nt * 16 + nl;
    for (int r = 0; r < 8; ++r) {
      int m = r + mb;
      if (n0 + m < a.N) {
        float x = c[r] + a.b[n];
        x = fmaxf(x, 0.f);
        x = fminf(x, 10.f);  // relu then clip(-10,10)
        atomicAdd(&a.pooled[(size_t)bb[m] * 512 + n], x);
      }
    }
  }
}

__global__ void k_final(const float* pooled, const float* cnt, float* out, int B) {
  int t = blockIdx.x * blockDim.x + threadIdx.x;
  if (t >= B * 512) return;
  int g = t / 512;
  out[t] = pooled[t] / fmaxf(cnt[g], 1.f);
}

// ---------------- host ----------------
struct GvpW { const float *wh, *ws_b, *ws_w, *wsv_b, *wsv_w, *wv; };

extern "C" void kernel_launch(void* const* d_in, const int* in_sizes, int n_in,
                              void* d_out, int out_size, void* d_ws, size_t ws_size,
                              hipStream_t stream) {
  (void)n_in; (void)ws_size;
  const float* edge_s = (const float*)d_in[0];
  const float* edge_v = (const float*)d_in[1];
  const float* params = (const float*)d_in[2];
  const int* atoms = (const int*)d_in[3];
  const int* edge_index = (const int*)d_in[4];
  const int* batch_id = (const int*)d_in[5];
  const int N = in_sizes[3];
  const int E = in_sizes[0] / 16;  // edge_s is [E,16]
  const int B = out_size / 512;

  // ---- flat param layout: jax tree_leaves order (dict keys sorted) ----
  size_t off = 0;
  auto take = [&](size_t n) { const float* p = params + off; off += n; return p; };
  auto takeGvp = [&](int si, int vi, int so, int vo) {
    GvpW g{}; int h = vi > vo ? vi : vo;
    if (vi) {                                         // sorted: wh, ws_b, ws_w, wsv_b, wsv_w, wv
      g.wh = take((size_t)h * vi);
      g.ws_b = take(so);
      g.ws_w = take((size_t)so * (si + h));
      if (vo) { g.wsv_b = take(vo); g.wsv_w = take((size_t)vo * so); g.wv = take((size_t)vo * h); }
    } else { g.ws_b = take(so); g.ws_w = take((size_t)so * si); }
    return g;
  };
  const float* embed = take(13 * 13);
  GvpW ff0[5], ff1[5], m0[5], m1[5], m2[5];
  const float *n0b[5], *n0g[5], *n1b[5], *n1g[5];
  for (int l = 0; l < 5; ++l) {       // layer dict sorted: ff0, ff1, m0, m1, m2, norm0, norm1
    ff0[l] = takeGvp(100, 16, 400, 32);
    ff1[l] = takeGvp(400, 32, 100, 16);
    m0[l]  = takeGvp(232, 33, 100, 16);
    m1[l]  = takeGvp(100, 16, 100, 16);
    m2[l]  = takeGvp(100, 16, 100, 16);
    n0b[l] = take(100); n0g[l] = take(100);   // LN dict sorted: b, g
    n1b[l] = take(100); n1g[l] = take(100);
  }
  GvpW we = takeGvp(16, 1, 32, 1);
  const float* web = take(16); const float* weg = take(16);
  GvpW wo = takeGvp(100, 16, 512, 0);
  const float* wob = take(100); const float* wog = take(100);
  GvpW wv = takeGvp(13, 0, 100, 0);
  const float* wvb = take(13); const float* wvg = take(13);

  // ---- workspace carve ----
  char* wsp = (char*)d_ws;
  size_t wof = 0;
  auto alloc = [&](size_t bytes) { size_t a = (wof + 255) & ~(size_t)255; void* p = wsp + a; wof = a + bytes; return p; };
  float* hsA = (float*)alloc((size_t)N * 100 * 4);
  float* hsB = (float*)alloc((size_t)N * 100 * 4);
  float* hvA = (float*)alloc((size_t)N * 48 * 4);
  float* hvB = (float*)alloc((size_t)N * 48 * 4);
  float* esb = (float*)alloc((size_t)E * 32 * 4);
  float* evb = (float*)alloc((size_t)E * 3 * 4);
  float* dhs = (float*)alloc((size_t)N * 100 * 4);
  float* dhv = (float*)alloc((size_t)N * 48 * 4);
  float* deg = (float*)alloc((size_t)N * 4);
  float* pooled = (float*)alloc((size_t)B * 512 * 4);
  float* bcnt = (float*)alloc((size_t)B * 4);

  auto pack = [&](const float* src, int rs, int cs, int rd, int cd) -> const _Float16* {
    _Float16* dst = (_Float16*)alloc((size_t)rd * cd * 2);
    int tot = rd * cd;
    k_pack<<<(tot + 255) / 256, 256, 0, stream>>>(src, dst, rs, cs, rd, cd);
    return dst;
  };
  const _Float16 *W0h[5], *G0h[5], *W1h[5], *G1h[5], *W2h[5], *G2h[5];
  const _Float16 *F0h[5], *FG0h[5], *F1h[5], *FG1h[5];
  for (int l = 0; l < 5; ++l) {
    W0h[l]  = pack(m0[l].ws_w, 100, 265, 112, 288);
    G0h[l]  = pack(m0[l].wsv_w, 16, 100, 16, 128);
    W1h[l]  = pack(m1[l].ws_w, 100, 116, 112, 128);
    G1h[l]  = pack(m1[l].wsv_w, 16, 100, 16, 128);
    W2h[l]  = pack(m2[l].ws_w, 100, 116, 112, 128);
    G2h[l]  = pack(m2[l].wsv_w, 16, 100, 16, 128);
    F0h[l]  = pack(ff0[l].ws_w, 400, 132, 400, 160);
    FG0h[l] = pack(ff0[l].wsv_w, 32, 400, 32, 416);
    F1h[l]  = pack(ff1[l].ws_w, 100, 432, 112, 448);
    FG1h[l] = pack(ff1[l].wsv_w, 16, 100, 16, 128);
  }
  const _Float16* WOh = pack(wo.ws_w, 512, 116, 512, 128);

  const int* esrc = edge_index;
  const int* edst = edge_index + E;

  k_node_init<<<(N + 255) / 256, 256, 0, stream>>>(atoms, embed, wvb, wvg, wv.ws_w, wv.ws_b, hsA, hvA, N);
  k_edge_init<<<(E + 255) / 256, 256, 0, stream>>>(edge_s, edge_v, web, weg, we.wh, we.ws_b,
                                                   we.ws_w, we.wsv_b, we.wsv_w, we.wv, esb, evb, E);
  hipMemsetAsync(deg, 0, (size_t)N * 4, stream);
  k_count<<<(E + 255) / 256, 256, 0, stream>>>(edst, deg, E);
  hipMemsetAsync(bcnt, 0, (size_t)B * 4, stream);
  k_count<<<(N + 255) / 256, 256, 0, stream>>>(batch_id, bcnt, N);

  float* hs_cur = hsA; float* hv_cur = hvA;
  float* hs_nxt = hsB; float* hv_nxt = hvB;
  for (int l = 0; l < 5; ++l) {
    hipMemsetAsync(dhs, 0, (size_t)N * 100 * 4, stream);
    hipMemsetAsync(dhv, 0, (size_t)N * 48 * 4, stream);
    ConvArgs ca{hs_cur, hv_cur, esb, evb, esrc, edst,
                W0h[l], m0[l].ws_b, m0[l].wh, m0[l].wv, G0h[l], m0[l].wsv_b,
                W1h[l], m1[l].ws_b, m1[l].wh, m1[l].wv, G1h[l], m1[l].wsv_b,
                W2h[l], m2[l].ws_b, m2[l].wh, m2[l].wv, G2h[l], m2[l].wsv_b,
                dhs, dhv, E};
    k_conv<<<(E + 15) / 16, 32, 0, stream>>>(ca);
    NodeArgs na{hs_cur, hv_cur, dhs, dhv, deg, n0b[l], n0g[l], n1b[l], n1g[l],
                F0h[l], ff0[l].ws_b, ff0[l].wh, ff0[l].wv, FG0h[l], ff0[l].wsv_b,
                F1h[l], ff1[l].ws_b, ff1[l].wh, ff1[l].wv, FG1h[l], ff1[l].wsv_b,
                hs_nxt, hv_nxt, N};
    k_node<<<(N + 15) / 16, 32, 0, stream>>>(na);
    float* ts = hs_cur; hs_cur = hs_nxt; hs_nxt = ts;
    float* tv = hv_cur; hv_cur = hv_nxt; hv_nxt = tv;
  }
  hipMemsetAsync(pooled, 0, (size_t)B * 512 * 4, stream);
  OutArgs oa{hs_cur, hv_cur, wob, wog, wo.wh, WOh, wo.ws_b, batch_id, pooled, N};
  k_out<<<(N + 15) / 16, 32, 0, stream>>>(oa);
  k_final<<<(B * 512 + 255) / 256, 256, 0, stream>>>(pooled, bcnt, (float*)d_out, B);
}
